// LocalInferenceModeling_24867860643935
// MI455X (gfx1250) — matrix operand order repaired and checked by reading, not compiled, not run on hardware
//
#include <hip/hip_runtime.h>
#include <hip/hip_bf16.h>

typedef __attribute__((ext_vector_type(16))) __bf16        v16bf;
typedef __attribute__((ext_vector_type(8)))  float         v8f;
typedef __attribute__((ext_vector_type(4)))  float         v4f;
typedef __attribute__((ext_vector_type(4)))  unsigned int  v4u;

#define LSTR 40   // padded LDS row stride in bf16 elems (80B: 16B-aligned, conflict-free)
#define Lr   1024
#define Dd   512
#define NB   16

// ---- helpers -------------------------------------------------------------
__device__ inline unsigned short f2bf(float f) {
  unsigned u = __float_as_uint(f);
  return (unsigned short)((u + 0x7FFFu + ((u >> 16) & 1u)) >> 16);  // RNE
}
__device__ inline unsigned int pack2(float lo, float hi) {
  return (unsigned int)f2bf(lo) | ((unsigned int)f2bf(hi) << 16);
}

struct Bits32 { v4u a, b; };

// A-operand 16x32 bf16 fragment: lane<16 row r holds K{0..7,16..23}; lane>=16 K{8..15,24..31}
__device__ inline v16bf frag_rowK(const unsigned short* lds, int row0, int lane) {
  int r = lane & 15, hi = (lane >> 4) & 1;
  const unsigned short* p = lds + (row0 + r) * LSTR + hi * 8;
  Bits32 x;
  x.a = *(const v4u*)(p);
  x.b = *(const v4u*)(p + 16);
  return __builtin_bit_cast(v16bf, x);
}
// B-operand 32x16 bf16 fragment: lane<16 col n holds K{0..7,16..23}; lane>=16 K{8..15,24..31}
__device__ inline v16bf frag_colK(const unsigned short* lds, int col0, int lane) {
  int n = lane & 15, hi = (lane >> 4) & 1;
  const unsigned short* p = lds + (col0 + n) * LSTR + hi * 8;
  Bits32 x;
  x.a = *(const v4u*)(p);
  x.b = *(const v4u*)(p + 16);
  return __builtin_bit_cast(v16bf, x);
}

#define WMMA_BF16(a, b, c) \
  __builtin_amdgcn_wmma_f32_16x16x32_bf16(false, (a), false, (b), (short)0, (c), false, false)

// Async global->LDS copy of 64 contiguous bytes (4 x b128). The 24-bit
// instruction offset is added to BOTH the LDS destination and the global
// source address (CDNA5 ISA), so one base pair suffices.
__device__ inline void async_copy64(unsigned lds_off, const void* g) {
  unsigned long long ga = (unsigned long long)g;
  asm volatile("global_load_async_to_lds_b128 %0, %1, off\n\t"
               "global_load_async_to_lds_b128 %0, %1, off offset:16\n\t"
               "global_load_async_to_lds_b128 %0, %1, off offset:32\n\t"
               "global_load_async_to_lds_b128 %0, %1, off offset:48"
               :: "v"(lds_off), "v"(ga) : "memory");
}
__device__ inline void wait_async0() {
  asm volatile("s_wait_asynccnt 0x0" ::: "memory");
}
__device__ inline unsigned lds_offset(const void* p) {
  // LDS aperture: generic addr[31:0] is the LDS byte offset
  return (unsigned)(unsigned long long)p;
}

// ---- kernel 1: score = A . B^T  (per batch 1024x1024x512) ----------------
// Staging uses the CDNA5 async copy engine (ASYNCcnt): raw fp32 slices are
// DMA'd global->LDS while WMMA crunches the previous slice; a short LDS->LDS
// pass converts to bf16 tiles for the fragment loaders.
__global__ __launch_bounds__(256) void score_kernel(const float* __restrict__ A,
                                                    const float* __restrict__ B,
                                                    float* __restrict__ S) {
  const int ta = blockIdx.x, tc = blockIdx.y, b = blockIdx.z;
  __shared__ __align__(16) float          Araw[2][128 * 32];   // 16 KB x2
  __shared__ __align__(16) float          Braw[2][128 * 32];   // 16 KB x2
  __shared__ __align__(16) unsigned short As[2][128 * LSTR];   // bf16 tiles
  __shared__ __align__(16) unsigned short Bs[2][128 * LSTR];
  const int t = threadIdx.x, lane = t & 31, w = t >> 5;
  const int wrow = (w & 3) * 32, wcol = (w >> 2) * 64;
  const int srow = t >> 1, shalf = t & 1;  // thread stages 64B of row srow
  const float* Ag = A + ((size_t)b * Lr + ta * 128 + srow) * Dd + shalf * 16;
  const float* Bg = B + ((size_t)b * Lr + tc * 128 + srow) * Dd + shalf * 16;
  const int rawIdx = srow * 32 + shalf * 16;

  v8f acc[2][4];
  const v8f vz = {0.f, 0.f, 0.f, 0.f, 0.f, 0.f, 0.f, 0.f};
#pragma unroll
  for (int i = 0; i < 2; ++i)
#pragma unroll
    for (int j = 0; j < 4; ++j) acc[i][j] = vz;

  auto issue = [&](int buf, int kb) {
    async_copy64(lds_offset(&Araw[buf][rawIdx]), Ag + kb);
    async_copy64(lds_offset(&Braw[buf][rawIdx]), Bg + kb);
  };
  auto convert = [&](int buf) {
    const v4f* ra = (const v4f*)&Araw[buf][rawIdx];
    const v4f* rb = (const v4f*)&Braw[buf][rawIdx];
    v4f a0 = ra[0], a1 = ra[1], a2 = ra[2], a3 = ra[3];
    v4f b0 = rb[0], b1 = rb[1], b2 = rb[2], b3 = rb[3];
    v4u pa0 = {pack2(a0[0], a0[1]), pack2(a0[2], a0[3]), pack2(a1[0], a1[1]), pack2(a1[2], a1[3])};
    v4u pa1 = {pack2(a2[0], a2[1]), pack2(a2[2], a2[3]), pack2(a3[0], a3[1]), pack2(a3[2], a3[3])};
    v4u pb0 = {pack2(b0[0], b0[1]), pack2(b0[2], b0[3]), pack2(b1[0], b1[1]), pack2(b1[2], b1[3])};
    v4u pb1 = {pack2(b2[0], b2[1]), pack2(b2[2], b2[3]), pack2(b3[0], b3[1]), pack2(b3[2], b3[3])};
    v4u* da = (v4u*)&As[buf][srow * LSTR + shalf * 16];
    v4u* db = (v4u*)&Bs[buf][srow * LSTR + shalf * 16];
    da[0] = pa0; da[1] = pa1;
    db[0] = pb0; db[1] = pb1;
  };

  // prologue: bring in first slice
  issue(0, 0);
  wait_async0();
  __syncthreads();
  convert(0);
  __syncthreads();

  int buf = 0;
  for (int kb = 0; kb < Dd; kb += 32) {
    const bool more = (kb + 32 < Dd);
    if (more) issue(buf ^ 1, kb + 32);   // async copy overlaps WMMA below

    v16bf afr[2], bfr[4];
    afr[0] = frag_rowK(As[buf], wrow, lane);
    afr[1] = frag_rowK(As[buf], wrow + 16, lane);
#pragma unroll
    for (int j = 0; j < 4; ++j) bfr[j] = frag_colK(Bs[buf], wcol + j * 16, lane);
#pragma unroll
    for (int i = 0; i < 2; ++i)
#pragma unroll
      for (int j = 0; j < 4; ++j) acc[i][j] = WMMA_BF16(afr[i], bfr[j], acc[i][j]);

    if (more) wait_async0();             // our async writes to raw[buf^1] landed
    __syncthreads();                     // everyone's landed; As[buf] reads done
    if (more) {
      convert(buf ^ 1);
      __syncthreads();
    }
    buf ^= 1;
  }

  const int r = lane & 15, hi = (lane >> 4) & 1;
#pragma unroll
  for (int i = 0; i < 2; ++i)
#pragma unroll
    for (int j = 0; j < 4; ++j) {
      int row = ta * 128 + wrow + i * 16 + hi * 8;
      int col = tc * 128 + wcol + j * 16 + r;
      float* sp = S + ((size_t)b * Lr + row) * Lr + col;
#pragma unroll
      for (int v = 0; v < 8; ++v) sp[(size_t)v * Lr] = acc[i][j][v];
    }
}

// ---- kernel 2: per-row max & sum(exp)  (one wave32 per row) --------------
__global__ __launch_bounds__(256) void row_stats_kernel(const float* __restrict__ S,
                                                        float* __restrict__ rmax,
                                                        float* __restrict__ rsum) {
  const int lane = threadIdx.x & 31, w = threadIdx.x >> 5;
  const int row = blockIdx.x * 8 + w;  // flat over NB*Lr rows
  const float* rp = S + (size_t)row * Lr;
  float m = -3.0e38f;
  for (int c = lane; c < Lr; c += 32) m = fmaxf(m, rp[c]);
#pragma unroll
  for (int o = 16; o; o >>= 1) m = fmaxf(m, __shfl_xor(m, o, 32));
  float s = 0.f;
  for (int c = lane; c < Lr; c += 32) s += __expf(rp[c] - m);
#pragma unroll
  for (int o = 16; o; o >>= 1) s += __shfl_xor(s, o, 32);
  if (lane == 0) { rmax[row] = m; rsum[row] = s; }
}

// ---- kernel 3: per-column max & sum(exp) (thread per column) -------------
__global__ __launch_bounds__(256) void col_stats_kernel(const float* __restrict__ S,
                                                        float* __restrict__ cmax,
                                                        float* __restrict__ csum) {
  const int c = blockIdx.x * 256 + threadIdx.x;
  const int b = blockIdx.y;
  const float* sp = S + (size_t)b * Lr * Lr + c;
  float m = -3.0e38f;
  for (int a = 0; a < Lr; ++a) m = fmaxf(m, sp[(size_t)a * Lr]);
  float s = 0.f;
  for (int a = 0; a < Lr; ++a) s += __expf(sp[(size_t)a * Lr] - m);
  cmax[b * Lr + c] = m;
  csum[b * Lr + c] = s;
}

// ---- kernel 4: A_t = softmax_row(score) . B ; write out_A sections -------
__global__ __launch_bounds__(256) void at_out_kernel(const float* __restrict__ A,
                                                     const float* __restrict__ B,
                                                     const float* __restrict__ S,
                                                     const float* __restrict__ rmaxv,
                                                     const float* __restrict__ rsumv,
                                                     float* __restrict__ outA) {
  const int td = blockIdx.x, ta = blockIdx.y, b = blockIdx.z;
  __shared__ __align__(16) unsigned short Ps[2][128 * LSTR];
  __shared__ __align__(16) unsigned short Bt[2][128 * LSTR];
  const int t = threadIdx.x, lane = t & 31, w = t >> 5;
  const int wrow = (w & 3) * 32, wcol = (w >> 2) * 64;
  const int srow = t >> 1, shalf = t & 1;
  const float* Sg = S + ((size_t)b * Lr + ta * 128 + srow) * Lr + shalf * 16;
  const float rmax = rmaxv[b * Lr + ta * 128 + srow];
  const int sc = t >> 3, se = t & 7;

  v8f acc[2][4];
  const v8f vz = {0.f, 0.f, 0.f, 0.f, 0.f, 0.f, 0.f, 0.f};
#pragma unroll
  for (int i = 0; i < 2; ++i)
#pragma unroll
    for (int j = 0; j < 4; ++j) acc[i][j] = vz;

  auto stage = [&](int buf, int kb) {
    // P = exp(score - rowmax) slice: [128 a][32 c], row-major
    if (kb + 32 < Lr) __builtin_prefetch(Sg + kb + 32, 0, 1);
    const v4f* gs = (const v4f*)(Sg + kb);
    float q[16];
#pragma unroll
    for (int i = 0; i < 4; ++i) {
      v4f x = gs[i];
      q[4 * i + 0] = x[0]; q[4 * i + 1] = x[1]; q[4 * i + 2] = x[2]; q[4 * i + 3] = x[3];
    }
    unsigned int pk[8];
#pragma unroll
    for (int i = 0; i < 8; ++i)
      pk[i] = pack2(__expf(q[2 * i] - rmax), __expf(q[2 * i + 1] - rmax));
    v4u* dp = (v4u*)&Ps[buf][srow * LSTR + shalf * 16];
    dp[0] = (v4u){pk[0], pk[1], pk[2], pk[3]};
    dp[1] = (v4u){pk[4], pk[5], pk[6], pk[7]};
    // B slice transposed into Bt[dd_local][c_local]
    const float* bg = B + ((size_t)b * Lr + kb + sc) * Dd + td * 128 + se * 16;
    const v4f* gb = (const v4f*)bg;
    float qb[16];
#pragma unroll
    for (int i = 0; i < 4; ++i) {
      v4f x = gb[i];
      qb[4 * i + 0] = x[0]; qb[4 * i + 1] = x[1]; qb[4 * i + 2] = x[2]; qb[4 * i + 3] = x[3];
    }
#pragma unroll
    for (int i = 0; i < 16; ++i) Bt[buf][(se * 16 + i) * LSTR + sc] = f2bf(qb[i]);
  };

  stage(0, 0);
  __syncthreads();
  int buf = 0;
  for (int kb = 0; kb < Lr; kb += 32) {
    if (kb + 32 < Lr) stage(buf ^ 1, kb + 32);
    v16bf afr[2], bfr[4];
    afr[0] = frag_rowK(Ps[buf], wrow, lane);
    afr[1] = frag_rowK(Ps[buf], wrow + 16, lane);
#pragma unroll
    for (int j = 0; j < 4; ++j) bfr[j] = frag_colK(Bt[buf], wcol + j * 16, lane);
#pragma unroll
    for (int i = 0; i < 2; ++i)
#pragma unroll
      for (int j = 0; j < 4; ++j) acc[i][j] = WMMA_BF16(afr[i], bfr[j], acc[i][j]);
    __syncthreads();
    buf ^= 1;
  }

  const int r = lane & 15, hi = (lane >> 4) & 1;
#pragma unroll
  for (int i = 0; i < 2; ++i)
#pragma unroll
    for (int j = 0; j < 4; ++j) {
      int dd = td * 128 + wcol + j * 16 + r;
      int rowbase = ta * 128 + wrow + i * 16 + hi * 8;
#pragma unroll
      for (int v = 0; v < 8; ++v) {
        int row = rowbase + v;
        float inv = 1.0f / rsumv[b * Lr + row];
        float tv = acc[i][j][v] * inv;
        float av = A[((size_t)b * Lr + row) * Dd + dd];
        float* op = outA + ((size_t)b * Lr + row) * 2048 + dd;
        op[0] = av; op[512] = tv; op[1024] = av - tv; op[1536] = av * tv;
      }
    }
}

// ---- kernel 5: B_t = softmax_col(score)^T . A ; write out_B sections -----
__global__ __launch_bounds__(256) void bt_out_kernel(const float* __restrict__ A,
                                                     const float* __restrict__ B,
                                                     const float* __restrict__ S,
                                                     const float* __restrict__ cmaxv,
                                                     const float* __restrict__ csumv,
                                                     float* __restrict__ outB) {
  const int td = blockIdx.x, tc = blockIdx.y, b = blockIdx.z;
  __shared__ __align__(16) unsigned short Pt[2][128 * LSTR];  // [c_local][a_local]
  __shared__ __align__(16) unsigned short At[2][128 * LSTR];  // [dd_local][a_local]
  const int t = threadIdx.x, lane = t & 31, w = t >> 5;
  const int wrow = (w & 3) * 32, wcol = (w >> 2) * 64;
  const int sa = t & 31, sc16 = t >> 5;  // P' staging
  const int sc = t >> 3, se = t & 7;     // A staging

  v8f acc[2][4];
  const v8f vz = {0.f, 0.f, 0.f, 0.f, 0.f, 0.f, 0.f, 0.f};
#pragma unroll
  for (int i = 0; i < 2; ++i)
#pragma unroll
    for (int j = 0; j < 4; ++j) acc[i][j] = vz;

  auto stage = [&](int buf, int kb) {
    // P' = exp(score - colmax) transposed into Pt[c_local][a_local]
    const float* sg = S + ((size_t)b * Lr + kb + sa) * Lr + tc * 128 + sc16 * 16;
    if (kb + 32 < Lr) __builtin_prefetch(sg + 32 * Lr, 0, 1);
    const v4f* gs = (const v4f*)sg;
    float q[16];
#pragma unroll
    for (int i = 0; i < 4; ++i) {
      v4f x = gs[i];
      q[4 * i + 0] = x[0]; q[4 * i + 1] = x[1]; q[4 * i + 2] = x[2]; q[4 * i + 3] = x[3];
    }
#pragma unroll
    for (int i = 0; i < 16; ++i) {
      float cm = cmaxv[b * Lr + tc * 128 + sc16 * 16 + i];
      Pt[buf][(sc16 * 16 + i) * LSTR + sa] = f2bf(__expf(q[i] - cm));
    }
    // A slice transposed into At[dd_local][a_local]
    const float* ag = A + ((size_t)b * Lr + kb + sc) * Dd + td * 128 + se * 16;
    const v4f* ga = (const v4f*)ag;
    float qa[16];
#pragma unroll
    for (int i = 0; i < 4; ++i) {
      v4f x = ga[i];
      qa[4 * i + 0] = x[0]; qa[4 * i + 1] = x[1]; qa[4 * i + 2] = x[2]; qa[4 * i + 3] = x[3];
    }
#pragma unroll
    for (int i = 0; i < 16; ++i) At[buf][(se * 16 + i) * LSTR + sc] = f2bf(qa[i]);
  };

  stage(0, 0);
  __syncthreads();
  int buf = 0;
  for (int kb = 0; kb < Lr; kb += 32) {
    if (kb + 32 < Lr) stage(buf ^ 1, kb + 32);
    v16bf afr[2], bfr[4];
    afr[0] = frag_rowK(Pt[buf], wrow, lane);
    afr[1] = frag_rowK(Pt[buf], wrow + 16, lane);
#pragma unroll
    for (int j = 0; j < 4; ++j) bfr[j] = frag_colK(At[buf], wcol + j * 16, lane);
#pragma unroll
    for (int i = 0; i < 2; ++i)
#pragma unroll
      for (int j = 0; j < 4; ++j) acc[i][j] = WMMA_BF16(afr[i], bfr[j], acc[i][j]);
    __syncthreads();
    buf ^= 1;
  }

  const int r = lane & 15, hi = (lane >> 4) & 1;
#pragma unroll
  for (int i = 0; i < 2; ++i)
#pragma unroll
    for (int j = 0; j < 4; ++j) {
      int dd = td * 128 + wcol + j * 16 + r;
      int cbase = tc * 128 + wrow + i * 16 + hi * 8;
#pragma unroll
      for (int v = 0; v < 8; ++v) {
        int c = cbase + v;
        float inv = 1.0f / csumv[b * Lr + c];
        float tv = acc[i][j][v] * inv;
        float bv = B[((size_t)b * Lr + c) * Dd + dd];
        float* op = outB + ((size_t)b * Lr + c) * 2048 + dd;
        op[0] = bv; op[512] = tv; op[1024] = bv - tv; op[1536] = bv * tv;
      }
    }
}

// ---- host launch ---------------------------------------------------------
extern "C" void kernel_launch(void* const* d_in, const int* in_sizes, int n_in,
                              void* d_out, int out_size, void* d_ws, size_t ws_size,
                              hipStream_t stream) {
  const float* A = (const float*)d_in[0];
  const float* B = (const float*)d_in[1];
  float* out = (float*)d_out;

  float* S    = (float*)d_ws;                 // 16*1024*1024 f32 = 64 MB (L2-resident)
  float* rmax = S + (size_t)NB * Lr * Lr;
  float* rsum = rmax + (size_t)NB * Lr;
  float* cmax = rsum + (size_t)NB * Lr;
  float* csum = cmax + (size_t)NB * Lr;

  float* outA = out;
  float* outB = out + (size_t)NB * Lr * 2048;

  score_kernel<<<dim3(8, 8, NB), 256, 0, stream>>>(A, B, S);
  row_stats_kernel<<<dim3(NB * Lr / 8), 256, 0, stream>>>(S, rmax, rsum);
  col_stats_kernel<<<dim3(Lr / 256, NB), 256, 0, stream>>>(S, cmax, csum);
  at_out_kernel<<<dim3(4, 8, NB), 256, 0, stream>>>(A, B, S, rmax, rsum, outA);
  bt_out_kernel<<<dim3(4, 8, NB), 256, 0, stream>>>(A, B, S, cmax, csum, outB);
}
